// MambaAdapter_90031104459358
// MI455X (gfx1250) — matrix operand order repaired and checked by reading, not compiled, not run on hardware
//
#include <hip/hip_runtime.h>
#include <hip/hip_bf16.h>

// ---------------- problem constants ----------------
#define DIMX    1024
#define DEPTH   4
#define DSTATE  16
#define DCONV   4
#define DINNER  2048
#define DTRANK  64
#define BSZ     4
#define SEQL    2048
#define NTOK    (BSZ * SEQL)      // 8192

typedef __attribute__((ext_vector_type(16))) _Float16 v16h;
typedef __attribute__((ext_vector_type(8)))  float    v8f;

// ---------------- WMMA GEMM: C[M,N] = A[M,K] * W[N,K]^T ----------------
// A, W are f16 row-major (K contiguous). C is f32.
// Block tile 128x128, K-step 32, 256 threads = 8 waves, each wave owns 32x64.
// Global->LDS staging uses CDNA5 async copies (ASYNCcnt) with double buffering.

#define BM 128
#define BN 128
#define BK 32
#define KSTR 40   // LDS row stride in halfs: 80 B -> 16B-aligned b128 chunks, even for b32 pair loads

__device__ __forceinline__ void wait_async0() {
    asm volatile("s_wait_asynccnt 0x0" ::: "memory");
}

// Per-lane async DMA of one [rows x 32-half] tile slice: 2 x 16B chunks per thread.
__device__ __forceinline__ void async_tile(const _Float16* __restrict__ g, int ld /*halfs*/,
                                           int rowBase, int rowMax, int k0,
                                           unsigned ldsBase /*bytes*/, int tid) {
#pragma unroll
    for (int i = 0; i < 2; ++i) {
        int idx = i * 256 + tid;
        int r = idx >> 2;        // 0..127 tile row
        int c = idx & 3;         // 16-byte chunk within the 64B row
        int gr = rowBase + r;
        if (gr < rowMax) {
            unsigned long long ga = (unsigned long long)(const void*)g +
                ((unsigned long long)gr * (unsigned)ld + (unsigned)k0) * 2ull +
                (unsigned)(c * 16);
            unsigned la = ldsBase + (unsigned)(r * (KSTR * 2) + c * 16);
            asm volatile("global_load_async_to_lds_b128 %0, %1, off"
                         :: "v"(la), "v"(ga) : "memory");
        }
    }
}

__device__ __forceinline__ v16h load_frag_A(const _Float16* lds, int rowBase, int lane) {
    // A fragment 16(M)x32(K), wave32 layout:
    //  lanes 0-15: M=lane, VGPR v(0..3): K={2v,2v+1}, v(4..7): K={16+2(v-4),...}
    //  lanes16-31: same M, K offset +8 within each half-group
    union { v16h v; unsigned u[8]; } f;
    const unsigned* p = (const unsigned*)lds;
    int r    = rowBase + (lane & 15);
    int khi  = (lane >> 4) * 8;
    int base = r * KSTR;
#pragma unroll
    for (int t = 0; t < 4; ++t)
        f.u[t] = p[(base + 2 * t + khi) >> 1];
#pragma unroll
    for (int t = 0; t < 4; ++t)
        f.u[4 + t] = p[(base + 16 + 2 * t + khi) >> 1];
    return f.v;
}

__device__ __forceinline__ v16h load_frag_B(const _Float16* lds, int nBase, int lane) {
    // B fragment 32(K)x16(N) built from LDS tile stored [N][K]:
    //  lanes 0-15: N=lane, K=0..7,16..23 ; lanes16-31: same N, K=8..15,24..31
    union { v16h v; unsigned u[8]; } f;
    const unsigned* p = (const unsigned*)lds;
    int n    = nBase + (lane & 15);
    int khi  = (lane >> 4) * 8;
    int base = n * KSTR + khi;
#pragma unroll
    for (int t = 0; t < 8; ++t)
        f.u[t] = p[(base + 2 * (t & 3) + (t >> 2) * 16) >> 1];
    return f.v;
}

template <bool SOFTPLUS_BIAS>
__global__ __launch_bounds__(256) void gemm_wmma(
    const _Float16* __restrict__ A, int lda,   // halfs
    const _Float16* __restrict__ W, int ldw,   // halfs
    const float* __restrict__ bias,            // [N] or nullptr
    float* __restrict__ C, int ldc,
    int M, int N, int K)
{
    __shared__ __attribute__((aligned(16))) _Float16 sA[2][BM * KSTR];
    __shared__ __attribute__((aligned(16))) _Float16 sB[2][BN * KSTR];

    const int tid  = threadIdx.x;
    const int lane = tid & 31;
    const int wave = tid >> 5;      // 0..7
    const int wy   = wave >> 1;     // 0..3  -> M offset 32*wy
    const int wx   = wave & 1;      // 0..1  -> N offset 64*wx
    const int blockM = blockIdx.y * BM;
    const int blockN = blockIdx.x * BN;

    const unsigned ldsA0 = (unsigned)(unsigned long long)(const void*)&sA[0][0];
    const unsigned ldsA1 = (unsigned)(unsigned long long)(const void*)&sA[1][0];
    const unsigned ldsB0 = (unsigned)(unsigned long long)(const void*)&sB[0][0];
    const unsigned ldsB1 = (unsigned)(unsigned long long)(const void*)&sB[1][0];

    v8f acc[2][4];
#pragma unroll
    for (int mi = 0; mi < 2; ++mi)
#pragma unroll
        for (int ni = 0; ni < 4; ++ni)
            acc[mi][ni] = (v8f){0.f, 0.f, 0.f, 0.f, 0.f, 0.f, 0.f, 0.f};

    // prologue: async-stage tile 0 into buffer 0
    async_tile(A, lda, blockM, M, 0, ldsA0, tid);
    async_tile(W, ldw, blockN, N, 0, ldsB0, tid);
    wait_async0();
    __syncthreads();

    int cur = 0;
    for (int k0 = 0; k0 < K; k0 += BK) {
        const bool more = (k0 + BK) < K;
        if (more) {
            // prefetch next K-tile into the other buffer while we compute
            async_tile(A, lda, blockM, M, k0 + BK, cur ? ldsA0 : ldsA1, tid);
            async_tile(W, ldw, blockN, N, k0 + BK, cur ? ldsB0 : ldsB1, tid);
        }

        const _Float16* bufA = sA[cur];
        const _Float16* bufB = sB[cur];

        v16h af[2], bf[4];
#pragma unroll
        for (int mi = 0; mi < 2; ++mi)
            af[mi] = load_frag_A(bufA, wy * 32 + mi * 16, lane);
#pragma unroll
        for (int ni = 0; ni < 4; ++ni)
            bf[ni] = load_frag_B(bufB, wx * 64 + ni * 16, lane);

#pragma unroll
        for (int mi = 0; mi < 2; ++mi)
#pragma unroll
            for (int ni = 0; ni < 4; ++ni)
                acc[mi][ni] = __builtin_amdgcn_wmma_f32_16x16x32_f16(
                    false, af[mi], false, bf[ni], (short)0, acc[mi][ni], false, false);

        if (more) {
            wait_async0();
            __syncthreads();
        }
        cur ^= 1;
    }

    // epilogue: D 16x16 f32 layout -> lanes 0-15: M=r, N=lane; lanes 16-31: M=8+r
#pragma unroll
    for (int mi = 0; mi < 2; ++mi) {
#pragma unroll
        for (int ni = 0; ni < 4; ++ni) {
            int col = blockN + wx * 64 + ni * 16 + (lane & 15);
            if (col >= N) continue;
            int rowBase = blockM + wy * 32 + mi * 16 + (lane >> 4) * 8;
            float bval = SOFTPLUS_BIAS ? bias[col] : 0.f;
#pragma unroll
            for (int r = 0; r < 8; ++r) {
                int rr = rowBase + r;
                if (rr < M) {
                    float v = acc[mi][ni][r];
                    if (SOFTPLUS_BIAS) {
                        v += bval;
                        v = (v > 20.f) ? v : log1pf(__expf(v));
                    }
                    C[(size_t)rr * ldc + col] = v;
                }
            }
        }
    }
}

// ---------------- f32 -> f16 convert ----------------
__global__ void cvt_f16_kernel(const float* __restrict__ in, _Float16* __restrict__ out, long n) {
    long i = (long)blockIdx.x * blockDim.x + threadIdx.x;
    long stride = (long)gridDim.x * blockDim.x;
    for (; i < n; i += stride) out[i] = (_Float16)in[i];
}

// ---------------- causal depthwise conv (k=4) + SiLU ----------------
// xz: [NTOK, 2*DINNER]; xi = cols [0, DINNER). Writes xc (f32) and xch (f16), [NTOK, DINNER].
__global__ void conv_silu_kernel(const float* __restrict__ xz,
                                 const float* __restrict__ cw,  // [DINNER,4]
                                 const float* __restrict__ cb,  // [DINNER]
                                 float* __restrict__ xc, _Float16* __restrict__ xch,
                                 long total)
{
    long i = (long)blockIdx.x * blockDim.x + threadIdx.x;
    long stride = (long)gridDim.x * blockDim.x;
    for (; i < total; i += stride) {
        int c = (int)(i & (DINNER - 1));
        long t = i >> 11;               // token = b*L + l
        int l = (int)(t & (SEQL - 1));
        float acc = cb[c];
#pragma unroll
        for (int j = 0; j < DCONV; ++j) {
            int ll = l - (DCONV - 1) + j;
            if (ll >= 0)
                acc += xz[(t - (DCONV - 1) + j) * (2 * DINNER) + c] * cw[c * DCONV + j];
        }
        float s = acc / (1.f + __expf(-acc));   // silu
        xc[i] = s;
        xch[i] = (_Float16)s;
    }
}

// ---------------- gather dt columns (first 64 of dbc) to f16 ----------------
__global__ void gather_dt_kernel(const float* __restrict__ dbc, _Float16* __restrict__ dth, long n) {
    long i = (long)blockIdx.x * blockDim.x + threadIdx.x;
    long stride = (long)gridDim.x * blockDim.x;
    for (; i < n; i += stride) {
        long t = i >> 6;
        int r = (int)(i & 63);
        dth[i] = (_Float16)dbc[t * (DTRANK + 2 * DSTATE) + r];
    }
}

// ---------------- selective scan ----------------
// One thread per (batch, channel). h[16] in registers, B/C broadcast via LDS.
// grid = BSZ * (DINNER/256) = 32 blocks of 256.
__global__ __launch_bounds__(256) void scan_kernel(
    const float* __restrict__ delta,   // [NTOK, DINNER]
    const float* __restrict__ xc,      // [NTOK, DINNER] (u)
    const float* __restrict__ dbc,     // [NTOK, 96]; B = cols 64..79, C = 80..95
    const float* __restrict__ xz,      // [NTOK, 2*DINNER]; z = cols DINNER..
    const float* __restrict__ A_log,   // [DINNER, DSTATE]
    const float* __restrict__ Dv,      // [DINNER]
    _Float16* __restrict__ yh)         // [NTOK, DINNER] gated output (f16)
{
    __shared__ float bc[2 * DSTATE];
    const int b = blockIdx.x >> 3;
    const int d = ((blockIdx.x & 7) << 8) + threadIdx.x;

    float Aa[DSTATE], h[DSTATE];
#pragma unroll
    for (int s = 0; s < DSTATE; ++s) {
        Aa[s] = -__expf(A_log[d * DSTATE + s]);
        h[s]  = 0.f;
    }
    const float Dd = Dv[d];

    for (int t = 0; t < SEQL; ++t) {
        const long token = (long)b * SEQL + t;
        if (threadIdx.x < 2 * DSTATE)
            bc[threadIdx.x] = dbc[token * (DTRANK + 2 * DSTATE) + DTRANK + threadIdx.x];
        __syncthreads();

        const long base = token * DINNER + d;
        const float dt = delta[base];
        const float u  = xc[base];
        const float du = dt * u;
        float y = 0.f;
#pragma unroll
        for (int s = 0; s < DSTATE; ++s) {
            float dA = __expf(dt * Aa[s]);
            h[s] = dA * h[s] + du * bc[s];
            y += h[s] * bc[DSTATE + s];
        }
        const float z = xz[token * (2 * DINNER) + DINNER + d];
        const float g = z / (1.f + __expf(-z));           // silu(z)
        yh[base] = (_Float16)((y + u * Dd) * g);
        __syncthreads();
    }
}

// ---------------- host orchestration ----------------
extern "C" void kernel_launch(void* const* d_in, const int* in_sizes, int n_in,
                              void* d_out, int out_size, void* d_ws, size_t ws_size,
                              hipStream_t stream) {
    (void)in_sizes; (void)n_in; (void)out_size; (void)ws_size;

    const float* x0    = (const float*)d_in[0];
    const float* inw   = (const float*)d_in[1];
    const float* convw = (const float*)d_in[2];
    const float* convb = (const float*)d_in[3];
    const float* xpw   = (const float*)d_in[4];
    const float* dtw   = (const float*)d_in[5];
    const float* dtb   = (const float*)d_in[6];
    const float* alog  = (const float*)d_in[7];
    const float* Dvec  = (const float*)d_in[8];
    const float* outw  = (const float*)d_in[9];

    char* base = (char*)d_ws;
    size_t off = 0;
    auto alloc = [&](size_t bytes) -> void* {
        off = (off + 255) & ~(size_t)255;
        void* p = base + off;
        off += bytes;
        return p;
    };

    _Float16* xh   = (_Float16*)alloc((size_t)NTOK * DIMX * 2);            // x in f16
    _Float16* wh   = (_Float16*)alloc((size_t)(2 * DINNER) * DIMX * 2);    // weight slot (max 4096x1024)
    float*    xz   = (float*)   alloc((size_t)NTOK * 2 * DINNER * 4);      // in_proj out
    float*    xc   = (float*)   alloc((size_t)NTOK * DINNER * 4);          // conv+silu out
    _Float16* xch  = (_Float16*)alloc((size_t)NTOK * DINNER * 2);
    float*    dbc  = (float*)   alloc((size_t)NTOK * (DTRANK + 2 * DSTATE) * 4);
    _Float16* dth  = (_Float16*)alloc((size_t)NTOK * DTRANK * 2);
    float*    dlt  = (float*)   alloc((size_t)NTOK * DINNER * 4);          // delta
    _Float16* yh   = (_Float16*)alloc((size_t)NTOK * DINNER * 2);          // gated y (f16)
    float*    xbuf = (float*)   alloc((size_t)NTOK * DIMX * 4);            // inter-layer x

    auto cvt = [&](const float* src, _Float16* dst, long n) {
        int blocks = (int)((n + 255) / 256);
        if (blocks > 32768) blocks = 32768;
        cvt_f16_kernel<<<blocks, 256, 0, stream>>>(src, dst, n);
    };

    const float* xcur = x0;
    for (int layer = 0; layer < DEPTH; ++layer) {
        const float* inw_l  = inw  + (size_t)layer * (2 * DINNER) * DIMX;
        const float* cw_l   = convw + (size_t)layer * DINNER * DCONV;
        const float* cb_l   = convb + (size_t)layer * DINNER;
        const float* xpw_l  = xpw  + (size_t)layer * (DTRANK + 2 * DSTATE) * DINNER;
        const float* dtw_l  = dtw  + (size_t)layer * DINNER * DTRANK;
        const float* dtb_l  = dtb  + (size_t)layer * DINNER;
        const float* alog_l = alog + (size_t)layer * DINNER * DSTATE;
        const float* D_l    = Dvec + (size_t)layer * DINNER;
        const float* outw_l = outw + (size_t)layer * DIMX * DINNER;

        // 1) in_proj: xz[8192,4096] = xh[8192,1024] * inw[4096,1024]^T
        cvt(xcur, xh, (long)NTOK * DIMX);
        cvt(inw_l, wh, (long)(2 * DINNER) * DIMX);
        gemm_wmma<false><<<dim3((2 * DINNER) / BN, NTOK / BM), 256, 0, stream>>>(
            xh, DIMX, wh, DIMX, nullptr, xz, 2 * DINNER, NTOK, 2 * DINNER, DIMX);

        // 2) causal depthwise conv + silu -> xc (f32) + xch (f16)
        conv_silu_kernel<<<16384, 256, 0, stream>>>(xz, cw_l, cb_l, xc, xch,
                                                    (long)NTOK * DINNER);

        // 3) x_proj: dbc[8192,96] = xch[8192,2048] * xpw[96,2048]^T
        cvt(xpw_l, wh, (long)(DTRANK + 2 * DSTATE) * DINNER);
        gemm_wmma<false><<<dim3(1, NTOK / BM), 256, 0, stream>>>(
            xch, DINNER, wh, DINNER, nullptr, dbc, DTRANK + 2 * DSTATE,
            NTOK, DTRANK + 2 * DSTATE, DINNER);

        // 4) dt_proj + softplus(bias+.): delta[8192,2048]
        gather_dt_kernel<<<2048, 256, 0, stream>>>(dbc, dth, (long)NTOK * DTRANK);
        cvt(dtw_l, wh, (long)DINNER * DTRANK);
        gemm_wmma<true><<<dim3(DINNER / BN, NTOK / BM), 256, 0, stream>>>(
            dth, DTRANK, wh, DTRANK, dtb_l, dlt, DINNER, NTOK, DINNER, DTRANK);

        // 5) selective scan + D skip + silu(z) gate -> yh (f16)
        scan_kernel<<<BSZ * (DINNER / 256), 256, 0, stream>>>(
            dlt, xc, dbc, xz, alog_l, D_l, yh);

        // 6) out_proj: x_next[8192,1024] = yh[8192,2048] * outw[1024,2048]^T
        cvt(outw_l, wh, (long)DIMX * DINNER);
        float* outp = (layer == DEPTH - 1) ? (float*)d_out : xbuf;
        gemm_wmma<false><<<dim3(DIMX / BN, NTOK / BM), 256, 0, stream>>>(
            yh, DINNER, wh, DINNER, nullptr, outp, DIMX, NTOK, DIMX, DINNER);

        xcur = outp;
    }
}
